// Qwen2MoeBlock_40286793237172
// MI455X (gfx1250) — matrix operand order repaired and checked by reading, not compiled, not run on hardware
//
#include <hip/hip_runtime.h>

#define B_ 2
#define S_ 1024
#define D_ 2048
#define H_ 16
#define DH_ 128
#define E_ 8
#define TOPK_ 2
#define I_ 1408
#define IS_ 5632
#define T_ (B_*S_)
#define EPS_ 1e-6f

typedef __attribute__((ext_vector_type(16))) __bf16 v16bf;
typedef __attribute__((ext_vector_type(8)))  float  v8f;
typedef __attribute__((ext_vector_type(4)))  float  f32x4;
typedef __attribute__((ext_vector_type(4)))  __bf16 bf16x4;

__device__ __forceinline__ __bf16 f2bf(float f) { return (__bf16)f; }

__device__ __forceinline__ bf16x4 cvt4(f32x4 v) {
  bf16x4 r;
#pragma unroll
  for (int i = 0; i < 4; i++) r[i] = (__bf16)v[i];
  return r;
}

__device__ __forceinline__ v8f zero8() {
  v8f z;
#pragma unroll
  for (int i = 0; i < 8; i++) z[i] = 0.f;
  return z;
}

__device__ __forceinline__ v8f wmma_bf16(v16bf a, v16bf b, v8f c) {
  return __builtin_amdgcn_wmma_f32_16x16x32_bf16(false, a, false, b, (short)0, c, false, false);
}

// ---------------- RMSNorm ----------------
__global__ __launch_bounds__(256) void rmsnorm_k(const float* __restrict__ x,
                                                 const float* __restrict__ w,
                                                 float* __restrict__ o) {
  __shared__ float red[8];
  const int row = blockIdx.x;
  const float* xr = x + (size_t)row * D_;
  float ss = 0.f;
  for (int d = threadIdx.x; d < D_; d += 256) { float v = xr[d]; ss += v * v; }
  for (int off = 16; off >= 1; off >>= 1) ss += __shfl_xor(ss, off, 32);
  if ((threadIdx.x & 31) == 0) red[threadIdx.x >> 5] = ss;
  __syncthreads();
  float tot = 0.f;
#pragma unroll
  for (int i = 0; i < 8; i++) tot += red[i];
  const float rinv = rsqrtf(tot / (float)D_ + EPS_);
  float* orow = o + (size_t)row * D_;
  for (int d = threadIdx.x; d < D_; d += 256) orow[d] = xr[d] * rinv * w[d];
}

// ---------------- Generic WMMA GEMM: C[M,N] = A[M,K] @ B[K,N] (+bias) ----------------
#define BM 64
#define BN 128
#define BK 32

__global__ __launch_bounds__(256) void wmma_gemm(const float* __restrict__ A,
                                                 const float* __restrict__ Bm,
                                                 const float* __restrict__ bias,
                                                 float* __restrict__ C,
                                                 int M, int N, int Kd) {
  __shared__ __bf16 As[BM][BK];
  __shared__ __bf16 Bst[BN][BK];   // transposed: [n][k]
  const int tid = threadIdx.x;
  const int lane = tid & 31;
  const int wid = tid >> 5;
  const int wm = wid >> 2;   // 0..1
  const int wn = wid & 3;    // 0..3
  const int hl = lane >> 4;
  const int l16 = lane & 15;
  const int m0 = blockIdx.y * BM;
  const int n0 = blockIdx.x * BN;

  v8f acc[2][2];
#pragma unroll
  for (int i = 0; i < 2; i++)
#pragma unroll
    for (int j = 0; j < 2; j++) acc[i][j] = zero8();

  for (int k0 = 0; k0 < Kd; k0 += BK) {
    // A tile: 64x32, float4 loads, packed bf16x4 LDS stores
#pragma unroll
    for (int i = 0; i < 2; i++) {
      int idx = tid + 256 * i;               // 0..511 groups of 4
      int r = idx >> 3, c4 = (idx & 7) * 4;
      f32x4 v = *(const f32x4*)&A[(size_t)(m0 + r) * Kd + k0 + c4];
      *(bf16x4*)&As[r][c4] = cvt4(v);
    }
    // B tile: 32x128, float4 loads, transposed scatter to LDS
#pragma unroll
    for (int i = 0; i < 4; i++) {
      int idx = tid + 256 * i;               // 0..1023 groups of 4
      int kr = idx >> 5, nc4 = (idx & 31) * 4;
      f32x4 v = *(const f32x4*)&Bm[(size_t)(k0 + kr) * N + n0 + nc4];
#pragma unroll
      for (int j = 0; j < 4; j++) Bst[nc4 + j][kr] = f2bf(v[j]);
    }
    __syncthreads();
    v16bf af[2], bf[2];
#pragma unroll
    for (int mt = 0; mt < 2; mt++) {
      const __bf16* p = &As[wm * 32 + mt * 16 + l16][0];
#pragma unroll
      for (int e = 0; e < 8; e++) af[mt][e] = p[hl * 8 + e];
#pragma unroll
      for (int e = 0; e < 8; e++) af[mt][8 + e] = p[16 + hl * 8 + e];
    }
#pragma unroll
    for (int nt = 0; nt < 2; nt++) {
      const __bf16* p = &Bst[wn * 32 + nt * 16 + l16][0];
#pragma unroll
      for (int e = 0; e < 16; e++) bf[nt][e] = p[(e & 8) * 2 + hl * 8 + (e & 7)];
    }
#pragma unroll
    for (int mt = 0; mt < 2; mt++)
#pragma unroll
      for (int nt = 0; nt < 2; nt++)
        acc[mt][nt] = wmma_bf16(af[mt], bf[nt], acc[mt][nt]);
    __syncthreads();
  }
  // hoisted bias (uniform branch, one load per column tile)
  float bv[2];
#pragma unroll
  for (int nt = 0; nt < 2; nt++)
    bv[nt] = bias ? bias[n0 + wn * 32 + nt * 16 + l16] : 0.f;
#pragma unroll
  for (int mt = 0; mt < 2; mt++)
#pragma unroll
    for (int nt = 0; nt < 2; nt++)
#pragma unroll
      for (int r = 0; r < 8; r++) {
        int mm = m0 + wm * 32 + mt * 16 + r + 8 * hl;
        int nn = n0 + wn * 32 + nt * 16 + l16;
        C[(size_t)mm * N + nn] = acc[mt][nt][r] + bv[nt];
      }
}

// ---------------- RoPE (paired halves, in-place, q & k) ----------------
__global__ __launch_bounds__(256) void rope_k(float* __restrict__ q, float* __restrict__ k,
                                              const float* __restrict__ sin_t,
                                              const float* __restrict__ cos_t,
                                              const int* __restrict__ pos) {
  int idx = blockIdx.x * blockDim.x + threadIdx.x;  // T*H*64
  if (idx >= T_ * H_ * 64) return;
  int d = idx & 63;
  int h = (idx >> 6) & (H_ - 1);
  int t = idx >> 10;
  int b = t / S_, s = t % S_;
  int p = pos[b * S_ + s];
  float c1 = cos_t[p * DH_ + d],      s1 = sin_t[p * DH_ + d];
  float c2 = cos_t[p * DH_ + d + 64], s2 = sin_t[p * DH_ + d + 64];
  size_t base = ((size_t)t * H_ + h) * DH_;
  float q1 = q[base + d], q2 = q[base + d + 64];
  q[base + d]      = q1 * c1 - q2 * s1;
  q[base + d + 64] = q2 * c2 + q1 * s2;
  float k1 = k[base + d], k2 = k[base + d + 64];
  k[base + d]      = k1 * c1 - k2 * s1;
  k[base + d + 64] = k2 * c2 + k1 * s2;
}

// ---------------- Flash attention: one wave per 16 queries ----------------
__global__ __launch_bounds__(32) void flash_attn_k(const float* __restrict__ q,
                                                   const float* __restrict__ k,
                                                   const float* __restrict__ v,
                                                   const int* __restrict__ amask,
                                                   float* __restrict__ o) {
  __shared__ __bf16 Ks[32][DH_];
  __shared__ __bf16 Vst[DH_][32];
  __shared__ __bf16 Ps[16][32];
  const int lane = threadIdx.x;
  const int hl = lane >> 4;
  const int l16 = lane & 15;
  const int qbase = blockIdx.x * 16;
  const int h = blockIdx.y;
  const int b = blockIdx.z;

  v16bf qf[4];
  {
    const float* qrow = q + (((size_t)(b * S_ + qbase + l16) * H_) + h) * DH_;
#pragma unroll
    for (int c = 0; c < 4; c++) {
      f32x4 lo0 = *(const f32x4*)&qrow[c * 32 + hl * 8];
      f32x4 lo1 = *(const f32x4*)&qrow[c * 32 + hl * 8 + 4];
      f32x4 hi0 = *(const f32x4*)&qrow[c * 32 + 16 + hl * 8];
      f32x4 hi1 = *(const f32x4*)&qrow[c * 32 + 16 + hl * 8 + 4];
#pragma unroll
      for (int e = 0; e < 4; e++) {
        qf[c][e]      = f2bf(lo0[e]);
        qf[c][4 + e]  = f2bf(lo1[e]);
        qf[c][8 + e]  = f2bf(hi0[e]);
        qf[c][12 + e] = f2bf(hi1[e]);
      }
    }
  }

  v8f Oacc[8];
#pragma unroll
  for (int c = 0; c < 8; c++) Oacc[c] = zero8();
  float mrow[8], lrow[8];
#pragma unroll
  for (int r = 0; r < 8; r++) { mrow[r] = -1e30f; lrow[r] = 0.f; }

  const float scale = 0.08838834764831845f;  // 1/sqrt(128)
  const int ntiles = (qbase + 15) / 32 + 1;
  for (int jt = 0; jt < ntiles; jt++) {
    const int kb = jt * 32;
    // K/V tiles: 32 keys x 128 dims, float4 loads (1024 groups / 32 lanes)
    for (int i = 0; i < 32; i++) {
      int e4 = lane + 32 * i;
      int key = e4 >> 5, d4 = (e4 & 31) * 4;
      size_t gidx = (((size_t)(b * S_ + kb + key) * H_) + h) * DH_ + d4;
      f32x4 kv = *(const f32x4*)&k[gidx];
      f32x4 vv = *(const f32x4*)&v[gidx];
      *(bf16x4*)&Ks[key][d4] = cvt4(kv);
#pragma unroll
      for (int j = 0; j < 4; j++) Vst[d4 + j][key] = f2bf(vv[j]);
    }
    __syncthreads();

    v8f c0 = zero8(), c1 = zero8();
#pragma unroll
    for (int c = 0; c < 4; c++) {
      v16bf b0, b1;
      const __bf16* pk0 = &Ks[l16][c * 32 + 8 * hl];
      const __bf16* pk1 = &Ks[16 + l16][c * 32 + 8 * hl];
#pragma unroll
      for (int e = 0; e < 16; e++) { b0[e] = pk0[(e & 8) * 2 + (e & 7)]; b1[e] = pk1[(e & 8) * 2 + (e & 7)]; }
      c0 = wmma_bf16(qf[c], b0, c0);
      c1 = wmma_bf16(qf[c], b1, c1);
    }

    float s0[8], s1[8];
#pragma unroll
    for (int r = 0; r < 8; r++) {
      int qi = qbase + r + 8 * hl;
      int k0i = kb + l16;
      int k1i = kb + 16 + l16;
      s0[r] = (k0i <= qi && amask[b * S_ + k0i] > 0) ? c0[r] * scale : -1e30f;
      s1[r] = (k1i <= qi && amask[b * S_ + k1i] > 0) ? c1[r] * scale : -1e30f;
    }
#pragma unroll
    for (int r = 0; r < 8; r++) {
      float mx = fmaxf(s0[r], s1[r]);
      for (int off = 8; off >= 1; off >>= 1) mx = fmaxf(mx, __shfl_xor(mx, off, 32));
      float mnew = fmaxf(mrow[r], mx);
      float alpha = __expf(mrow[r] - mnew);
      float p0 = __expf(s0[r] - mnew);
      float p1 = __expf(s1[r] - mnew);
      float rs = p0 + p1;
      for (int off = 8; off >= 1; off >>= 1) rs += __shfl_xor(rs, off, 32);
      lrow[r] = lrow[r] * alpha + rs;
      mrow[r] = mnew;
#pragma unroll
      for (int c = 0; c < 8; c++) Oacc[c][r] *= alpha;
      Ps[r + 8 * hl][l16] = f2bf(p0);
      Ps[r + 8 * hl][16 + l16] = f2bf(p1);
    }
    __syncthreads();

    v16bf pa;
    {
      const __bf16* pp = &Ps[l16][0];
#pragma unroll
      for (int e = 0; e < 8; e++) pa[e] = pp[hl * 8 + e];
#pragma unroll
      for (int e = 0; e < 8; e++) pa[8 + e] = pp[16 + hl * 8 + e];
    }
#pragma unroll
    for (int c = 0; c < 8; c++) {
      v16bf vb;
      const __bf16* pv = &Vst[c * 16 + l16][8 * hl];
#pragma unroll
      for (int e = 0; e < 16; e++) vb[e] = pv[(e & 8) * 2 + (e & 7)];
      Oacc[c] = wmma_bf16(pa, vb, Oacc[c]);
    }
    __syncthreads();
  }
#pragma unroll
  for (int c = 0; c < 8; c++)
#pragma unroll
    for (int r = 0; r < 8; r++) {
      int m = r + 8 * hl;
      float val = Oacc[c][r] / lrow[r];
      o[(((size_t)(b * S_ + qbase + m) * H_) + h) * DH_ + c * 16 + l16] = val;
    }
}

// ---------------- elementwise helpers ----------------
__global__ __launch_bounds__(256) void add_k(const float* __restrict__ a,
                                             const float* __restrict__ b,
                                             float* __restrict__ c, int n) {
  int i = blockIdx.x * blockDim.x + threadIdx.x;
  if (i < n) c[i] = a[i] + b[i];
}

__global__ __launch_bounds__(256) void silu_mul_k(float* __restrict__ g,
                                                  const float* __restrict__ u, int n) {
  int i = blockIdx.x * blockDim.x + threadIdx.x;
  if (i < n) { float x = g[i]; g[i] = x / (1.f + __expf(-x)) * u[i]; }
}

// ---------------- router + shared gate ----------------
__global__ __launch_bounds__(32) void router_k(const float* __restrict__ x,
                                               const float* __restrict__ rw,
                                               const float* __restrict__ gw,
                                               int* __restrict__ topi,
                                               float* __restrict__ topv,
                                               float* __restrict__ sgate) {
  const int t = blockIdx.x, lane = threadIdx.x;
  const float* xr = x + (size_t)t * D_;
  float acc[E_];
#pragma unroll
  for (int e = 0; e < E_; e++) acc[e] = 0.f;
  float gacc = 0.f;
  for (int d = lane; d < D_; d += 32) {
    float xv = xr[d];
#pragma unroll
    for (int e = 0; e < E_; e++) acc[e] += xv * rw[d * E_ + e];
    gacc += xv * gw[d];
  }
#pragma unroll
  for (int e = 0; e < E_; e++)
    for (int off = 16; off >= 1; off >>= 1) acc[e] += __shfl_xor(acc[e], off, 32);
  for (int off = 16; off >= 1; off >>= 1) gacc += __shfl_xor(gacc, off, 32);
  if (lane == 0) {
    float mx = acc[0];
#pragma unroll
    for (int e = 1; e < E_; e++) mx = fmaxf(mx, acc[e]);
    float p[E_], sum = 0.f;
#pragma unroll
    for (int e = 0; e < E_; e++) { p[e] = __expf(acc[e] - mx); sum += p[e]; }
#pragma unroll
    for (int e = 0; e < E_; e++) p[e] /= sum;
    int i1 = 0;
#pragma unroll
    for (int e = 1; e < E_; e++) if (p[e] > p[i1]) i1 = e;
    int i2 = (i1 == 0) ? 1 : 0;
#pragma unroll
    for (int e = 0; e < E_; e++) if (e != i1 && p[e] > p[i2]) i2 = e;
    topi[t * 2] = i1;     topv[t * 2] = p[i1];
    topi[t * 2 + 1] = i2; topv[t * 2 + 1] = p[i2];
    sgate[t] = 1.f / (1.f + __expf(-gacc));
  }
}

// ---------------- deterministic per-expert list build ----------------
__global__ void build_lists_k(const int* __restrict__ topi, const float* __restrict__ topv,
                              int* __restrict__ cnt, int* __restrict__ off,
                              int* __restrict__ list_tok, float* __restrict__ list_w,
                              int* __restrict__ slot_of) {
  int c[E_], cur[E_];
  for (int e = 0; e < E_; e++) c[e] = 0;
  for (int i = 0; i < T_ * TOPK_; i++) c[topi[i]]++;
  int run = 0;
  for (int e = 0; e < E_; e++) { off[e] = run; cur[e] = run; cnt[e] = c[e]; run += c[e]; }
  for (int t = 0; t < T_; t++)
    for (int s = 0; s < TOPK_; s++) {
      int e = topi[t * 2 + s];
      int p = cur[e]++;
      list_tok[p] = t; list_w[p] = topv[t * 2 + s]; slot_of[t * 2 + s] = p;
    }
}

// ---------------- MoE gate+up grouped GEMM (gathered A) + SiLU*up*w epilogue ----------------
__global__ __launch_bounds__(256) void moe_gateup_k(const float* __restrict__ X,
                                                    const float* __restrict__ egw,
                                                    const float* __restrict__ euw,
                                                    const int* __restrict__ cnt,
                                                    const int* __restrict__ off,
                                                    const int* __restrict__ list_tok,
                                                    const float* __restrict__ list_w,
                                                    float* __restrict__ hh) {
  const int ex = blockIdx.z;
  const int count = cnt[ex];
  const int m0 = blockIdx.y * BM;
  if (m0 >= count) return;
  const int base = off[ex];
  __shared__ __bf16 As[BM][BK];
  __shared__ __bf16 Bg[BN][BK];
  __shared__ __bf16 Bu[BN][BK];
  __shared__ int stok[BM];
  __shared__ float sw[BM];
  const int tid = threadIdx.x;
  const int lane = tid & 31;
  const int wid = tid >> 5;
  const int wm = wid >> 2, wn = wid & 3;
  const int hl = lane >> 4, l16 = lane & 15;
  const int n0 = blockIdx.x * BN;
  if (tid < BM) {
    int r = m0 + tid;
    if (r < count) { stok[tid] = list_tok[base + r]; sw[tid] = list_w[base + r]; }
    else { stok[tid] = -1; sw[tid] = 0.f; }
  }
  __syncthreads();
  const float* Bge = egw + (size_t)ex * D_ * I_;
  const float* Bue = euw + (size_t)ex * D_ * I_;

  v8f ag[2][2], au[2][2];
#pragma unroll
  for (int i = 0; i < 2; i++)
#pragma unroll
    for (int j = 0; j < 2; j++) { ag[i][j] = zero8(); au[i][j] = zero8(); }

  for (int k0 = 0; k0 < D_; k0 += BK) {
#pragma unroll
    for (int i = 0; i < 2; i++) {
      int idx = tid + 256 * i;
      int r = idx >> 3, c4 = (idx & 7) * 4;
      int tok = stok[r];
      f32x4 v;
      if (tok >= 0) v = *(const f32x4*)&X[(size_t)tok * D_ + k0 + c4];
      else { v[0] = 0.f; v[1] = 0.f; v[2] = 0.f; v[3] = 0.f; }
      *(bf16x4*)&As[r][c4] = cvt4(v);
    }
#pragma unroll
    for (int i = 0; i < 4; i++) {
      int idx = tid + 256 * i;
      int kr = idx >> 5, nc4 = (idx & 31) * 4;
      size_t bi = (size_t)(k0 + kr) * I_ + n0 + nc4;
      f32x4 vg = *(const f32x4*)&Bge[bi];
      f32x4 vu = *(const f32x4*)&Bue[bi];
#pragma unroll
      for (int j = 0; j < 4; j++) { Bg[nc4 + j][kr] = f2bf(vg[j]); Bu[nc4 + j][kr] = f2bf(vu[j]); }
    }
    __syncthreads();
    v16bf af[2], bg[2], bu[2];
#pragma unroll
    for (int mt = 0; mt < 2; mt++) {
      const __bf16* p = &As[wm * 32 + mt * 16 + l16][0];
#pragma unroll
      for (int e = 0; e < 8; e++) af[mt][e] = p[hl * 8 + e];
#pragma unroll
      for (int e = 0; e < 8; e++) af[mt][8 + e] = p[16 + hl * 8 + e];
    }
#pragma unroll
    for (int nt = 0; nt < 2; nt++) {
      const __bf16* pg = &Bg[wn * 32 + nt * 16 + l16][0];
      const __bf16* pu = &Bu[wn * 32 + nt * 16 + l16][0];
#pragma unroll
      for (int e = 0; e < 16; e++) { bg[nt][e] = pg[(e & 8) * 2 + hl * 8 + (e & 7)]; bu[nt][e] = pu[(e & 8) * 2 + hl * 8 + (e & 7)]; }
    }
#pragma unroll
    for (int mt = 0; mt < 2; mt++)
#pragma unroll
      for (int nt = 0; nt < 2; nt++) {
        ag[mt][nt] = wmma_bf16(af[mt], bg[nt], ag[mt][nt]);
        au[mt][nt] = wmma_bf16(af[mt], bu[nt], au[mt][nt]);
      }
    __syncthreads();
  }
#pragma unroll
  for (int mt = 0; mt < 2; mt++)
#pragma unroll
    for (int nt = 0; nt < 2; nt++)
#pragma unroll
      for (int r = 0; r < 8; r++) {
        int rl = wm * 32 + mt * 16 + r + 8 * hl;
        if (m0 + rl < count) {
          float gv = ag[mt][nt][r];
          float uv = au[mt][nt][r];
          float val = gv / (1.f + __expf(-gv)) * uv * sw[rl];
          hh[(size_t)(base + m0 + rl) * I_ + n0 + wn * 32 + nt * 16 + l16] = val;
        }
      }
}

// ---------------- MoE down grouped GEMM into per-slot buffer ----------------
__global__ __launch_bounds__(256) void moe_down_k(const float* __restrict__ hh,
                                                  const float* __restrict__ edw,
                                                  const int* __restrict__ cnt,
                                                  const int* __restrict__ off,
                                                  float* __restrict__ moe_part) {
  const int ex = blockIdx.z;
  const int count = cnt[ex];
  const int m0 = blockIdx.y * BM;
  if (m0 >= count) return;
  const int base = off[ex];
  __shared__ __bf16 As[BM][BK];
  __shared__ __bf16 Bst[BN][BK];
  const int tid = threadIdx.x;
  const int lane = tid & 31;
  const int wid = tid >> 5;
  const int wm = wid >> 2, wn = wid & 3;
  const int hl = lane >> 4, l16 = lane & 15;
  const int n0 = blockIdx.x * BN;
  const float* Be = edw + (size_t)ex * I_ * D_;

  v8f acc[2][2];
#pragma unroll
  for (int i = 0; i < 2; i++)
#pragma unroll
    for (int j = 0; j < 2; j++) acc[i][j] = zero8();

  for (int k0 = 0; k0 < I_; k0 += BK) {
#pragma unroll
    for (int i = 0; i < 2; i++) {
      int idx = tid + 256 * i;
      int r = idx >> 3, c4 = (idx & 7) * 4;
      f32x4 v;
      if (m0 + r < count) v = *(const f32x4*)&hh[(size_t)(base + m0 + r) * I_ + k0 + c4];
      else { v[0] = 0.f; v[1] = 0.f; v[2] = 0.f; v[3] = 0.f; }
      *(bf16x4*)&As[r][c4] = cvt4(v);
    }
#pragma unroll
    for (int i = 0; i < 4; i++) {
      int idx = tid + 256 * i;
      int kr = idx >> 5, nc4 = (idx & 31) * 4;
      f32x4 v = *(const f32x4*)&Be[(size_t)(k0 + kr) * D_ + n0 + nc4];
#pragma unroll
      for (int j = 0; j < 4; j++) Bst[nc4 + j][kr] = f2bf(v[j]);
    }
    __syncthreads();
    v16bf af[2], bf[2];
#pragma unroll
    for (int mt = 0; mt < 2; mt++) {
      const __bf16* p = &As[wm * 32 + mt * 16 + l16][0];
#pragma unroll
      for (int e = 0; e < 8; e++) af[mt][e] = p[hl * 8 + e];
#pragma unroll
      for (int e = 0; e < 8; e++) af[mt][8 + e] = p[16 + hl * 8 + e];
    }
#pragma unroll
    for (int nt = 0; nt < 2; nt++) {
      const __bf16* p = &Bst[wn * 32 + nt * 16 + l16][0];
#pragma unroll
      for (int e = 0; e < 16; e++) bf[nt][e] = p[(e & 8) * 2 + hl * 8 + (e & 7)];
    }
#pragma unroll
    for (int mt = 0; mt < 2; mt++)
#pragma unroll
      for (int nt = 0; nt < 2; nt++)
        acc[mt][nt] = wmma_bf16(af[mt], bf[nt], acc[mt][nt]);
    __syncthreads();
  }
#pragma unroll
  for (int mt = 0; mt < 2; mt++)
#pragma unroll
    for (int nt = 0; nt < 2; nt++)
#pragma unroll
      for (int r = 0; r < 8; r++) {
        int rl = wm * 32 + mt * 16 + r + 8 * hl;
        if (m0 + rl < count)
          moe_part[(size_t)(base + m0 + rl) * D_ + n0 + wn * 32 + nt * 16 + l16] =
              acc[mt][nt][r];
      }
}

// ---------------- final combine ----------------
__global__ __launch_bounds__(256) void combine_k(const float* __restrict__ h,
                                                 const float* __restrict__ moe_part,
                                                 const int* __restrict__ slot_of,
                                                 const float* __restrict__ sgate,
                                                 const float* __restrict__ sh,
                                                 float* __restrict__ out) {
  int i = blockIdx.x * blockDim.x + threadIdx.x;
  if (i >= T_ * D_) return;
  int t = i / D_;
  int n = i - t * D_;
  float moe = moe_part[(size_t)slot_of[t * 2] * D_ + n] +
              moe_part[(size_t)slot_of[t * 2 + 1] * D_ + n];
  out[i] = h[i] + moe + sgate[t] * sh[i];
}

extern "C" void kernel_launch(void* const* d_in, const int* in_sizes, int n_in,
                              void* d_out, int out_size, void* d_ws, size_t ws_size,
                              hipStream_t stream) {
  (void)in_sizes; (void)n_in; (void)out_size; (void)ws_size;
  const float* x      = (const float*)d_in[0];
  const int*   pos    = (const int*)d_in[1];
  const int*   amask  = (const int*)d_in[2];
  const float* sin_t  = (const float*)d_in[3];
  const float* cos_t  = (const float*)d_in[4];
  const float* ln1    = (const float*)d_in[5];
  const float* ln2    = (const float*)d_in[6];
  const float* q_w    = (const float*)d_in[7];
  const float* q_b    = (const float*)d_in[8];
  const float* k_w    = (const float*)d_in[9];
  const float* k_b    = (const float*)d_in[10];
  const float* v_w    = (const float*)d_in[11];
  const float* v_b    = (const float*)d_in[12];
  const float* o_w    = (const float*)d_in[13];
  const float* r_w    = (const float*)d_in[14];
  const float* eg_w   = (const float*)d_in[15];
  const float* eu_w   = (const float*)d_in[16];
  const float* ed_w   = (const float*)d_in[17];
  const float* sg_w   = (const float*)d_in[18];
  const float* su_w   = (const float*)d_in[19];
  const float* sd_w   = (const float*)d_in[20];
  const float* gate_w = (const float*)d_in[21];
  float* out = (float*)d_out;

  char* wp = (char*)d_ws;
  auto alloc_f = [&](size_t n) { float* p = (float*)wp; wp += ((n + 3) & ~(size_t)3) * sizeof(float); return p; };
  auto alloc_i = [&](size_t n) { int* p = (int*)wp; wp += ((n + 3) & ~(size_t)3) * sizeof(int); return p; };

  float* h1    = alloc_f((size_t)T_ * D_);
  float* qb    = alloc_f((size_t)T_ * D_);
  float* kb    = alloc_f((size_t)T_ * D_);
  float* vb    = alloc_f((size_t)T_ * D_);
  float* attn  = alloc_f((size_t)T_ * D_);
  float* aproj = alloc_f((size_t)T_ * D_);
  float* hbuf  = alloc_f((size_t)T_ * D_);
  float* h2    = alloc_f((size_t)T_ * D_);
  float* g1    = alloc_f((size_t)T_ * IS_);
  float* g2    = alloc_f((size_t)T_ * IS_);
  float* shout = alloc_f((size_t)T_ * D_);
  float* sgate = alloc_f(T_);
  float* topv  = alloc_f((size_t)T_ * TOPK_);
  float* lw    = alloc_f((size_t)T_ * TOPK_);
  float* hh    = alloc_f((size_t)T_ * TOPK_ * I_);
  float* moep  = alloc_f((size_t)T_ * TOPK_ * D_);
  int* topi    = alloc_i((size_t)T_ * TOPK_);
  int* slot_of = alloc_i((size_t)T_ * TOPK_);
  int* ltok    = alloc_i((size_t)T_ * TOPK_);
  int* cnt     = alloc_i(E_);
  int* off     = alloc_i(E_);

  // 1. pre-attention RMSNorm
  rmsnorm_k<<<T_, 256, 0, stream>>>(x, ln1, h1);
  // 2. QKV projections (WMMA GEMMs)
  wmma_gemm<<<dim3(D_ / BN, T_ / BM), 256, 0, stream>>>(h1, q_w, q_b, qb, T_, D_, D_);
  wmma_gemm<<<dim3(D_ / BN, T_ / BM), 256, 0, stream>>>(h1, k_w, k_b, kb, T_, D_, D_);
  wmma_gemm<<<dim3(D_ / BN, T_ / BM), 256, 0, stream>>>(h1, v_w, v_b, vb, T_, D_, D_);
  // 3. RoPE in place
  rope_k<<<(T_ * H_ * 64 + 255) / 256, 256, 0, stream>>>(qb, kb, sin_t, cos_t, pos);
  // 4. causal flash attention (WMMA)
  flash_attn_k<<<dim3(S_ / 16, H_, B_), 32, 0, stream>>>(qb, kb, vb, amask, attn);
  // 5. output projection + residual
  wmma_gemm<<<dim3(D_ / BN, T_ / BM), 256, 0, stream>>>(attn, o_w, nullptr, aproj, T_, D_, D_);
  add_k<<<(T_ * D_ + 255) / 256, 256, 0, stream>>>(x, aproj, hbuf, T_ * D_);
  // 6. post-attention RMSNorm
  rmsnorm_k<<<T_, 256, 0, stream>>>(hbuf, ln2, h2);
  // 7. router (softmax top-2) + shared gate
  router_k<<<T_, 32, 0, stream>>>(h2, r_w, gate_w, topi, topv, sgate);
  build_lists_k<<<1, 1, 0, stream>>>(topi, topv, cnt, off, ltok, lw, slot_of);
  // 8. MoE experts (grouped WMMA GEMMs)
  moe_gateup_k<<<dim3(I_ / BN, T_ / BM, E_), 256, 0, stream>>>(h2, eg_w, eu_w, cnt, off,
                                                              ltok, lw, hh);
  moe_down_k<<<dim3(D_ / BN, T_ / BM, E_), 256, 0, stream>>>(hh, ed_w, cnt, off, moep);
  // 9. shared expert
  wmma_gemm<<<dim3(IS_ / BN, T_ / BM), 256, 0, stream>>>(h2, sg_w, nullptr, g1, T_, IS_, D_);
  wmma_gemm<<<dim3(IS_ / BN, T_ / BM), 256, 0, stream>>>(h2, su_w, nullptr, g2, T_, IS_, D_);
  silu_mul_k<<<(T_ * IS_ + 255) / 256, 256, 0, stream>>>(g1, g2, T_ * IS_);
  wmma_gemm<<<dim3(D_ / BN, T_ / BM), 256, 0, stream>>>(g1, sd_w, nullptr, shout, T_, D_, IS_);
  // 10. combine: out = h + moe + sigmoid(gate)*shared
  combine_k<<<(T_ * D_ + 255) / 256, 256, 0, stream>>>(hbuf, moep, slot_of, sgate, shout, out);
}
